// Classifier_85839216378002
// MI455X (gfx1250) — compile-verified, matrix operand order checked by reading, not run on hardware
//
#include <hip/hip_runtime.h>
#include <hip/hip_bf16.h>
#include <math.h>

// Problem sizes (fixed by the reference)
#define IOn  64
#define Hdim 1024
#define Sdim 512
#define Nb   64
#define KFULL (IOn * Hdim)   // 65536
#define KSPLIT 16
#define BP_LDK 136           // padded LDS row stride (elements) -> bank-conflict free

typedef __attribute__((ext_vector_type(16))) __bf16 v16bf;
typedef __attribute__((ext_vector_type(8)))  __bf16 v8bf;
typedef __attribute__((ext_vector_type(8)))  float  v8f;

// ---------------- helpers ----------------

__device__ __forceinline__ unsigned short f2bf(float x) {
  unsigned int u = __float_as_uint(x);
  u += 0x7FFFu + ((u >> 16) & 1u);      // round-to-nearest-even
  return (unsigned short)(u >> 16);
}

// Load a 16x32 bf16 WMMA A/B fragment from row-major memory (global or LDS).
// ISA 16-bit layout: lane L holds row (L&15); K elements split as
// [kbase + sel*8 .. +7] and [kbase + 16 + sel*8 .. +7], sel = L>>4.
// Both groups are 16B-contiguous -> b128 loads.
__device__ __forceinline__ v16bf load_frag(const unsigned short* base,
                                           int stride, int kbase) {
  const int lane = threadIdx.x & 31;
  const int row  = lane & 15;
  const int sel  = lane >> 4;
  const unsigned short* p = base + (size_t)row * (size_t)stride + kbase + sel * 8;
  v8bf lo = *reinterpret_cast<const v8bf*>(p);
  v8bf hi = *reinterpret_cast<const v8bf*>(p + 16);
  v16bf f;
#pragma unroll
  for (int i = 0; i < 8; ++i) { f[i] = lo[i]; f[i + 8] = hi[i]; }
  return f;
}

__device__ __forceinline__ v8f wmma_bf16(v16bf a, v16bf b, v8f c) {
  return __builtin_amdgcn_wmma_f32_16x16x32_bf16(false, a, false, b,
                                                 (short)0, c, false, false);
}

// Cooperative staging of a 64-row x 128-col bf16 B panel: global -> regs -> LDS.
// 64*128 elems = 1024 uint4; NTHR threads carry PER = 1024/NTHR each.
template <int NTHR, int PER>
__device__ __forceinline__ void stage_to_regs(const unsigned short* B0, int strideB,
                                              int k0, uint4* r, int tid) {
#pragma unroll
  for (int it = 0; it < PER; ++it) {
    int i = it * NTHR + tid;
    int row = i >> 4, cc = i & 15;
    r[it] = *reinterpret_cast<const uint4*>(B0 + (size_t)row * strideB + k0 + cc * 8);
  }
}
template <int NTHR, int PER>
__device__ __forceinline__ void regs_to_lds(unsigned short* Bs, const uint4* r, int tid) {
#pragma unroll
  for (int it = 0; it < PER; ++it) {
    int i = it * NTHR + tid;
    int row = i >> 4, cc = i & 15;
    *reinterpret_cast<uint4*>(Bs + row * BP_LDK + cc * 8) = r[it];
  }
}

// ---------------- prep kernels ----------------

__global__ void k_convert_states(const float* __restrict__ st,
                                 unsigned short* __restrict__ st_bf,
                                 unsigned short* __restrict__ stT_bf) {
  size_t idx = (size_t)blockIdx.x * blockDim.x + threadIdx.x;
  const size_t total = (size_t)Nb * Sdim * Hdim;
  if (idx >= total) return;
  unsigned short b = f2bf(st[idx]);
  st_bf[idx] = b;
  size_t n = idx / ((size_t)Sdim * Hdim);
  size_t rem = idx - n * (size_t)Sdim * Hdim;
  size_t s = rem / Hdim, h = rem - s * Hdim;
  stT_bf[n * (size_t)Hdim * Sdim + h * Sdim + s] = b;
}

__global__ void k_convert_os(const float* __restrict__ os,
                             unsigned short* __restrict__ os_bf) {
  int idx = blockIdx.x * blockDim.x + threadIdx.x;
  if (idx < IOn * Hdim) os_bf[idx] = f2bf(os[idx]);
}

// WoT[j][k] = Wo[k][j] (top half of Wo only), bf16
__global__ void k_build_woT(const float* __restrict__ Wo,
                            unsigned short* __restrict__ woT) {
  int idx = blockIdx.x * blockDim.x + threadIdx.x;
  if (idx >= Hdim * Hdim) return;
  int j = idx >> 10, k = idx & 1023;
  woT[idx] = f2bf(Wo[(size_t)k * Hdim + j]);
}

// WcT[c][m] = Wc[m][c], bf16
__global__ void k_build_wcT(const float* __restrict__ Wc,
                            unsigned short* __restrict__ wcT) {
  size_t idx = (size_t)blockIdx.x * blockDim.x + threadIdx.x;
  if (idx >= (size_t)IOn * KFULL) return;
  size_t c = idx >> 16, m = idx & 65535;
  wcT[idx] = f2bf(Wc[m * IOn + c]);
}

// constT[i][j] = bo[j] + sum_k os[i][k] * Wo[H+k][j]   (batch-invariant term)
__global__ void k_const(const float* __restrict__ os, const float* __restrict__ Wo,
                        const float* __restrict__ bo, float* __restrict__ constT) {
  int idx = blockIdx.x * blockDim.x + threadIdx.x;
  if (idx >= IOn * Hdim) return;
  int i = idx >> 10, j = idx & 1023;
  float a = bo[j];
  const float* w = Wo + (size_t)Hdim * Hdim + j;
  const float* o = os + (size_t)i * Hdim;
  for (int k = 0; k < Hdim; ++k) a += o[k] * w[(size_t)k * Hdim];
  constT[idx] = a;
}

// ---------------- GEMM 1: scores[n][s][io] = states[n] . os^T ----------------

__global__ __launch_bounds__(128) void k_attn(const unsigned short* __restrict__ st_bf,
                                              const unsigned short* __restrict__ os_bf,
                                              float* __restrict__ scores) {
  __shared__ unsigned short Bs[64 * BP_LDK];
  const int n = blockIdx.y;
  const int tid = threadIdx.x;
  const int wave = tid >> 5;
  const int mtile = blockIdx.x * 4 + wave;   // s-tile 0..31
  const unsigned short* A =
      st_bf + (size_t)n * Sdim * Hdim + (size_t)(mtile * 16) * Hdim;
  uint4 rg[8];
  stage_to_regs<128, 8>(os_bf, Hdim, 0, rg, tid);
  regs_to_lds<128, 8>(Bs, rg, tid);
  __syncthreads();
  v8f acc[4] = {};
  const int nch = Hdim / 128;
  for (int c = 0; c < nch; ++c) {
    if (c + 1 < nch) stage_to_regs<128, 8>(os_bf, Hdim, (c + 1) * 128, rg, tid);
    const int kb = c * 128;
#pragma unroll
    for (int kk = 0; kk < 128; kk += 32) {
      v16bf a = load_frag(A, Hdim, kb + kk);
#pragma unroll
      for (int nt = 0; nt < 4; ++nt) {
        v16bf b = load_frag(Bs + nt * 16 * BP_LDK, BP_LDK, kk);
        acc[nt] = wmma_bf16(a, b, acc[nt]);
      }
    }
    __syncthreads();
    if (c + 1 < nch) { regs_to_lds<128, 8>(Bs, rg, tid); __syncthreads(); }
  }
  const int lane = tid & 31;
  const int nl = lane & 15, moff = (lane >> 4) * 8;
  float* out = scores + (size_t)n * Sdim * IOn + (size_t)(mtile * 16) * IOn;
#pragma unroll
  for (int nt = 0; nt < 4; ++nt)
#pragma unroll
    for (int v = 0; v < 8; ++v)
      out[(size_t)(moff + v) * IOn + nt * 16 + nl] = acc[nt][v];
}

// ---------------- softmax over s, write P[n][io][s] (bf16) ----------------

__global__ void k_softmax(const float* __restrict__ scores,
                          unsigned short* __restrict__ P) {
  const int n = blockIdx.x, io = threadIdx.x;
  const float* col = scores + (size_t)n * Sdim * IOn + io;
  float m = -1e30f;
  for (int s = 0; s < Sdim; ++s) m = fmaxf(m, col[(size_t)s * IOn]);
  float sum = 0.f;
  for (int s = 0; s < Sdim; ++s) sum += __expf(col[(size_t)s * IOn] - m);
  float inv = 1.0f / sum;
  unsigned short* prow = P + (size_t)n * IOn * Sdim + (size_t)io * Sdim;
  for (int s = 0; s < Sdim; ++s)
    prow[s] = f2bf(__expf(col[(size_t)s * IOn] - m) * inv);
}

// ------- GEMM 2: mix^T[h][io] = statesT[n] . P^T; store mix[io][h] -------

__global__ __launch_bounds__(128) void k_mix(const unsigned short* __restrict__ stT_bf,
                                             const unsigned short* __restrict__ P,
                                             unsigned short* __restrict__ mix_bf) {
  __shared__ unsigned short Bs[64 * BP_LDK];
  const int n = blockIdx.y;
  const int tid = threadIdx.x;
  const int wave = tid >> 5;
  const int mtile = blockIdx.x * 4 + wave;   // h-tile 0..63
  const unsigned short* A =
      stT_bf + (size_t)n * Hdim * Sdim + (size_t)(mtile * 16) * Sdim;
  const unsigned short* B0 = P + (size_t)n * IOn * Sdim;
  uint4 rg[8];
  stage_to_regs<128, 8>(B0, Sdim, 0, rg, tid);
  regs_to_lds<128, 8>(Bs, rg, tid);
  __syncthreads();
  v8f acc[4] = {};
  const int nch = Sdim / 128;
  for (int c = 0; c < nch; ++c) {
    if (c + 1 < nch) stage_to_regs<128, 8>(B0, Sdim, (c + 1) * 128, rg, tid);
    const int kb = c * 128;
#pragma unroll
    for (int kk = 0; kk < 128; kk += 32) {
      v16bf a = load_frag(A, Sdim, kb + kk);
#pragma unroll
      for (int nt = 0; nt < 4; ++nt) {
        v16bf b = load_frag(Bs + nt * 16 * BP_LDK, BP_LDK, kk);
        acc[nt] = wmma_bf16(a, b, acc[nt]);
      }
    }
    __syncthreads();
    if (c + 1 < nch) { regs_to_lds<128, 8>(Bs, rg, tid); __syncthreads(); }
  }
  const int lane = tid & 31;
  const int nl = lane & 15, moff = (lane >> 4) * 8;
#pragma unroll
  for (int nt = 0; nt < 4; ++nt)
#pragma unroll
    for (int v = 0; v < 8; ++v) {
      int io = nt * 16 + nl;
      int h = mtile * 16 + moff + v;
      mix_bf[(size_t)n * IOn * Hdim + (size_t)io * Hdim + h] = f2bf(acc[nt][v]);
    }
}

// ---- GEMM 3: o = mix . WoT^T + const, tanh, store Tmat[n][i*H+j] bf16 ----

__global__ __launch_bounds__(128) void k_out_tanh(const unsigned short* __restrict__ mix_bf,
                                                  const unsigned short* __restrict__ woT,
                                                  const float* __restrict__ constT,
                                                  unsigned short* __restrict__ Tmat) {
  __shared__ unsigned short Bs[64 * BP_LDK];
  const int n = blockIdx.y;
  const int jchunk = blockIdx.x;             // 0..15 (64 j per chunk)
  const int tid = threadIdx.x;
  const int wave = tid >> 5;                 // m-tile 0..3 (io)
  const unsigned short* A =
      mix_bf + (size_t)n * IOn * Hdim + (size_t)(wave * 16) * Hdim;
  const unsigned short* B0 = woT + (size_t)(jchunk * 64) * Hdim;
  uint4 rg[8];
  stage_to_regs<128, 8>(B0, Hdim, 0, rg, tid);
  regs_to_lds<128, 8>(Bs, rg, tid);
  __syncthreads();
  v8f acc[4] = {};
  const int nch = Hdim / 128;
  for (int c = 0; c < nch; ++c) {
    if (c + 1 < nch) stage_to_regs<128, 8>(B0, Hdim, (c + 1) * 128, rg, tid);
    const int kb = c * 128;
#pragma unroll
    for (int kk = 0; kk < 128; kk += 32) {
      v16bf a = load_frag(A, Hdim, kb + kk);
#pragma unroll
      for (int nt = 0; nt < 4; ++nt) {
        v16bf b = load_frag(Bs + nt * 16 * BP_LDK, BP_LDK, kk);
        acc[nt] = wmma_bf16(a, b, acc[nt]);
      }
    }
    __syncthreads();
    if (c + 1 < nch) { regs_to_lds<128, 8>(Bs, rg, tid); __syncthreads(); }
  }
  const int lane = tid & 31;
  const int nl = lane & 15, moff = (lane >> 4) * 8;
#pragma unroll
  for (int nt = 0; nt < 4; ++nt)
#pragma unroll
    for (int v = 0; v < 8; ++v) {
      int i = wave * 16 + moff + v;
      int j = jchunk * 64 + nt * 16 + nl;
      float o = acc[nt][v] + constT[(size_t)i * Hdim + j];
      Tmat[(size_t)n * KFULL + (size_t)i * Hdim + j] = f2bf(tanhf(o));
    }
}

// ---- GEMM 4: logits = Tmat[64,65536] . Wc + bc, K-split partials ----

__global__ __launch_bounds__(512) void k_final_partial(const unsigned short* __restrict__ Tmat,
                                                       const unsigned short* __restrict__ wcT,
                                                       float* __restrict__ partial) {
  __shared__ unsigned short Bs[64 * BP_LDK];
  const int blk = blockIdx.x;                 // k-split 0..15
  const int tid = threadIdx.x;
  const int wave = tid >> 5;                  // 16 waves: 4x4 output tiles
  const int mtile = wave >> 2, ntile = wave & 3;
  const int kspan = KFULL / KSPLIT;           // 4096
  const int k0 = blk * kspan;
  const unsigned short* A = Tmat + (size_t)(mtile * 16) * KFULL;
  uint4 rg[2];
  stage_to_regs<512, 2>(wcT, KFULL, k0, rg, tid);
  regs_to_lds<512, 2>(Bs, rg, tid);
  __syncthreads();
  v8f acc = {};
  const int nch = kspan / 128;                // 32
  for (int c = 0; c < nch; ++c) {
    if (c + 1 < nch) stage_to_regs<512, 2>(wcT, KFULL, k0 + (c + 1) * 128, rg, tid);
    const int kb = k0 + c * 128;
#pragma unroll
    for (int kk = 0; kk < 128; kk += 32) {
      v16bf a = load_frag(A, KFULL, kb + kk);
      v16bf b = load_frag(Bs + ntile * 16 * BP_LDK, BP_LDK, kk);
      acc = wmma_bf16(a, b, acc);
    }
    __syncthreads();
    if (c + 1 < nch) { regs_to_lds<512, 2>(Bs, rg, tid); __syncthreads(); }
  }
  const int lane = tid & 31;
  const int nl = lane & 15, moff = (lane >> 4) * 8;
  float* out = partial + (size_t)blk * Nb * IOn;
#pragma unroll
  for (int v = 0; v < 8; ++v)
    out[(size_t)(mtile * 16 + moff + v) * IOn + ntile * 16 + nl] = acc[v];
}

__global__ void k_final_reduce(const float* __restrict__ partial,
                               const float* __restrict__ bc,
                               float* __restrict__ out) {
  int idx = blockIdx.x * blockDim.x + threadIdx.x;
  if (idx >= Nb * IOn) return;
  float a = bc[idx & (IOn - 1)];
  for (int p = 0; p < KSPLIT; ++p) a += partial[(size_t)p * Nb * IOn + idx];
  out[idx] = a;
}

// ---------------- launch ----------------

extern "C" void kernel_launch(void* const* d_in, const int* in_sizes, int n_in,
                              void* d_out, int out_size, void* d_ws, size_t ws_size,
                              hipStream_t stream) {
  const float* states = (const float*)d_in[0];   // [64,512,1024]
  const float* os     = (const float*)d_in[1];   // [64,1024]
  const float* Wo     = (const float*)d_in[2];   // [2048,1024]
  const float* bo     = (const float*)d_in[3];   // [1024]
  const float* Wc     = (const float*)d_in[4];   // [65536,64]
  const float* bc     = (const float*)d_in[5];   // [64]
  float* out = (float*)d_out;                    // [64,64]

  char* w = (char*)d_ws;
  size_t off = 0;
  auto take = [&](size_t bytes) -> void* {
    void* p = w + off;
    off += (bytes + 255) & ~(size_t)255;
    return p;
  };
  unsigned short* st_bf  = (unsigned short*)take((size_t)Nb * Sdim * Hdim * 2);
  unsigned short* stT_bf = (unsigned short*)take((size_t)Nb * Sdim * Hdim * 2);
  unsigned short* os_bf  = (unsigned short*)take((size_t)IOn * Hdim * 2);
  unsigned short* woT    = (unsigned short*)take((size_t)Hdim * Hdim * 2);
  unsigned short* wcT    = (unsigned short*)take((size_t)IOn * KFULL * 2);
  float*          constT = (float*)take((size_t)IOn * Hdim * 4);
  float*          scores = (float*)take((size_t)Nb * Sdim * IOn * 4);
  unsigned short* P      = (unsigned short*)take((size_t)Nb * IOn * Sdim * 2);
  unsigned short* mix_bf = (unsigned short*)take((size_t)Nb * IOn * Hdim * 2);
  unsigned short* Tmat   = (unsigned short*)take((size_t)Nb * KFULL * 2);
  float*          part   = (float*)take((size_t)KSPLIT * Nb * IOn * 4);
  (void)ws_size; (void)in_sizes; (void)n_in; (void)out_size;

  const size_t stTotal = (size_t)Nb * Sdim * Hdim;
  k_convert_states<<<(unsigned)((stTotal + 255) / 256), 256, 0, stream>>>(states, st_bf, stT_bf);
  k_convert_os<<<(IOn * Hdim + 255) / 256, 256, 0, stream>>>(os, os_bf);
  k_build_woT<<<(Hdim * Hdim + 255) / 256, 256, 0, stream>>>(Wo, woT);
  k_build_wcT<<<(unsigned)(((size_t)IOn * KFULL + 255) / 256), 256, 0, stream>>>(Wc, wcT);
  k_const<<<(IOn * Hdim + 255) / 256, 256, 0, stream>>>(os, Wo, bo, constT);

  k_attn<<<dim3(8, Nb), 128, 0, stream>>>(st_bf, os_bf, scores);
  k_softmax<<<Nb, IOn, 0, stream>>>(scores, P);
  k_mix<<<dim3(16, Nb), 128, 0, stream>>>(stT_bf, P, mix_bf);
  k_out_tanh<<<dim3(16, Nb), 128, 0, stream>>>(mix_bf, woT, constT, Tmat);
  k_final_partial<<<KSPLIT, 512, 0, stream>>>(Tmat, wcT, part);
  k_final_reduce<<<(Nb * IOn + 255) / 256, 256, 0, stream>>>(part, bc, out);
}
